// sampleLSTM_AR_17360257811156
// MI455X (gfx1250) — compile-verified, matrix operand order checked by reading, not run on hardware
//
#include <hip/hip_runtime.h>
#include <hip/hip_bf16.h>
#include <math.h>

typedef __attribute__((ext_vector_type(16))) _Float16 v16h;
typedef __attribute__((ext_vector_type(8)))  _Float16 v8h;
typedef __attribute__((ext_vector_type(8)))  float    v8f;

#define BATCH 64
#define TLEN  271
#define HID   512
#define G4H   2048
#define HOR   16
#define WIN   256        // TLEN - HOR + 1
#define NTILES 128       // G4H / 16
#define KB0   16         // HID / 32
#define KB1   32         // (2*HID) / 32

// ---- workspace layout (bytes) ----
static constexpr size_t OFF_W0P   = 0;                         // 2 MiB  f16 packed Whh0
static constexpr size_t OFF_W1P   = (size_t)2 << 20;           // 4 MiB  f16 packed [Wih1;Whh1]
static constexpr size_t OFF_BIAS1 = (size_t)6 << 20;           // 8 KiB  f32 bih1+bhh1
static constexpr size_t OFF_W0COL = ((size_t)6 << 20) + 8192;  // 8 KiB  f32 Wih0[:,0]
static constexpr size_t OFF_PRED  = ((size_t)6 << 20) + 16384; // 4 KiB  f32 pred[t][b]
static constexpr size_t OFF_FLAG  = ((size_t)6 << 20) + 20480; // flags
static constexpr size_t OFF_Z0    = (size_t)8 << 20;           // 271*2048*64*4 f32 [pos][g][b]

__device__ __forceinline__ float sigmf(float x) { return 1.0f / (1.0f + __expf(-x)); }

// A operand (16x32, MxK, f16) from LDS h buffer laid out [b][k] row-major.
// lanes 0-15: M=l,  halves = K[k0..k0+7]    then K[k0+16..k0+23]
// lanes 16-31: M=l, halves = K[k0+8..k0+15] then K[k0+24..k0+31]
__device__ __forceinline__ v16h load_a_lds(const _Float16* buf, int mbase, int k0) {
  const int lane = threadIdx.x & 31;
  const int l = lane & 15, hs = lane >> 4;
  const _Float16* p = buf + (size_t)(mbase + l) * HID + k0 + hs * 8;
  v8h lo = *(const v8h*)p;
  v8h hi = *(const v8h*)(p + 16);
  v16h a;
#pragma unroll
  for (int e = 0; e < 8; ++e) { a[e] = lo[e]; a[e + 8] = hi[e]; }
  return a;
}

// B operand (32x16, KxN, f16) pre-packed so each lane reads 16 contiguous halves.
__device__ __forceinline__ v16h load_b_pack(const _Float16* wp, int nt, int kb, int kbn) {
  const int lane = threadIdx.x & 31;
  return *(const v16h*)(wp + (((size_t)nt * kbn + kb) * 32 + lane) * 16);
}

// ---------------- prep: pack weights to WMMA B layout, fold biases ----------------
__global__ void lstm_prep_pack_kernel(const float* __restrict__ Wih0,
                                      const float* __restrict__ Whh0,
                                      const float* __restrict__ Wih1,
                                      const float* __restrict__ Whh1,
                                      const float* __restrict__ bih1,
                                      const float* __restrict__ bhh1,
                                      _Float16* __restrict__ w0p,
                                      _Float16* __restrict__ w1p,
                                      float* __restrict__ bias1,
                                      float* __restrict__ w0col,
                                      int* __restrict__ flags) {
  const int total0 = NTILES * KB0 * 32 * 16;  // 1,048,576
  const int total1 = NTILES * KB1 * 32 * 16;  // 2,097,152
  for (int idx = blockIdx.x * blockDim.x + threadIdx.x; idx < total1;
       idx += gridDim.x * blockDim.x) {
    const int e    = idx & 15;
    const int lane = (idx >> 4) & 31;
    const int blk  = idx >> 9;  // nt*KBn + kb
    const int nn   = lane & 15;
    const int hsl  = lane >> 4;
    const int kk   = (e < 8) ? (hsl * 8 + e) : (16 + hsl * 8 + (e - 8));
    {  // layer-1 concatenated [Wih1 ; Whh1], K = 1024
      const int kb = blk % KB1, nt = blk / KB1;
      const int g = nt * 16 + nn;
      const int k = kb * 32 + kk;
      const float v = (k < HID) ? Wih1[(size_t)g * HID + k]
                                : Whh1[(size_t)g * HID + (k - HID)];
      w1p[idx] = (_Float16)v;
    }
    if (idx < total0) {  // layer-0 Whh0, K = 512
      const int kb = blk % KB0, nt = blk / KB0;
      const int g = nt * 16 + nn;
      const int k = kb * 32 + kk;
      w0p[idx] = (_Float16)Whh0[(size_t)g * HID + k];
    }
    if (idx < G4H) {
      bias1[idx] = bih1[idx] + bhh1[idx];
      w0col[idx] = Wih0[(size_t)idx * 8];
    }
    if (idx < HOR) flags[idx] = 0;
  }
}

// ---------------- prep: layer-0 input projection for all absolute positions ----------------
// Z0[pos][g][b] = bih0[g]+bhh0[g] + sum_i feat[b][pos][i] * Wih0[g][i]
__global__ void lstm_zproj_kernel(const float* __restrict__ feat,
                                  const float* __restrict__ Wih0,
                                  const float* __restrict__ bih0,
                                  const float* __restrict__ bhh0,
                                  float* __restrict__ Z0) {
  const size_t total = (size_t)TLEN * G4H * BATCH;
  for (size_t idx = (size_t)blockIdx.x * blockDim.x + threadIdx.x; idx < total;
       idx += (size_t)gridDim.x * blockDim.x) {
    const int b   = (int)(idx & (BATCH - 1));
    const int g   = (int)((idx >> 6) & (G4H - 1));
    const int pos = (int)(idx >> 17);
    float a = bih0[g] + bhh0[g];
    const float* x = feat + ((size_t)b * TLEN + pos) * 8;
    const float* w = Wih0 + (size_t)g * 8;
#pragma unroll
    for (int i = 0; i < 8; ++i) a += x[i] * w[i];
    Z0[idx] = a;
  }
}

// ---------------- main: one persistent block per horizon window ----------------
// Double-buffered h0/h1 in LDS: a pair's gates are consumed immediately after its
// GEMM (no gate tiles live across barriers -> no spills); 2 barriers per step.
__global__ __launch_bounds__(1024)
void lstm_window_kernel(const float* __restrict__ feat,
                        const float* __restrict__ Wout,
                        const float* __restrict__ bout,
                        const _Float16* __restrict__ w0p,
                        const _Float16* __restrict__ w1p,
                        const float* __restrict__ bias1,
                        const float* __restrict__ w0col,
                        const float* __restrict__ Z0,
                        float* __restrict__ predbuf,
                        int* __restrict__ flags,
                        float* __restrict__ out) {
  __shared__ __align__(16) _Float16 h0A[BATCH * HID];  // 64 KiB each
  __shared__ __align__(16) _Float16 h0B[BATCH * HID];
  __shared__ __align__(16) _Float16 h1A[BATCH * HID];
  __shared__ __align__(16) _Float16 h1B[BATCH * HID];
  __shared__ float predsh[BATCH];
  __shared__ float predacc[BATCH];

  const int t    = blockIdx.x;   // window / horizon index
  const int tid  = threadIdx.x;
  const int wave = tid >> 5;
  const int lane = tid & 31;
  const int n    = lane & 15;    // N within 16x16 tile
  const int hs   = lane >> 4;    // lane-half selects M rows r / r+8

  for (int i = tid; i < BATCH * HID; i += 1024) {
    h0A[i] = (_Float16)0.f;
    h0B[i] = (_Float16)0.f;
    h1A[i] = (_Float16)0.f;
    h1B[i] = (_Float16)0.f;
  }
  if (tid < BATCH) predsh[tid] = 0.f;
  __syncthreads();

  // wave owns 4 (batch-tile, hidden-unit-tile) pairs: 128 pairs / 32 waves
  int mt[4], ht[4];
#pragma unroll
  for (int q = 0; q < 4; ++q) {
    const int pair = wave * 4 + q;
    mt[q] = pair & 3;    // batch tile 0..3
    ht[q] = pair >> 2;   // hidden-unit tile 0..31
  }

  v8f c0[4], c1[4];
#pragma unroll
  for (int q = 0; q < 4; ++q)
#pragma unroll
    for (int r = 0; r < 8; ++r) { c0[q][r] = 0.f; c1[q][r] = 0.f; }

  const int firstTail = WIN - t;  // step index where pos reaches WIN (only if t>0)
  const _Float16* h1last = h1A;

#pragma unroll 1
  for (int s = 0; s < WIN; ++s) {
    const int pos = t + s;
    if (t > 0 && s == firstTail) {
      if (tid == 0) {
        while (__hip_atomic_load(&flags[t - 1], __ATOMIC_ACQUIRE,
                                 __HIP_MEMORY_SCOPE_AGENT) == 0)
          __builtin_amdgcn_s_sleep(2);
      }
      __syncthreads();
      if (tid < BATCH) predsh[tid] = predbuf[(size_t)(t - 1) * BATCH + tid];
      __syncthreads();
    }
    const bool tail = (pos >= WIN);
    const int par = s & 1;
    const _Float16* h0r = par ? h0B : h0A;
    _Float16*       h0w = par ? h0A : h0B;
    const _Float16* h1r = par ? h1B : h1A;
    _Float16*       h1w = par ? h1A : h1B;
    h1last = h1w;

    // ======== layer 0: g = Z0[pos] (+tail corr) + h0_prev @ Whh0^T ========
#pragma unroll 1
    for (int q = 0; q < 4; ++q) {
      v8f acc[4];
#pragma unroll
      for (int gi = 0; gi < 4; ++gi) {
        const int g = gi * HID + ht[q] * 16 + n;
        acc[gi] = *(const v8f*)(Z0 + ((size_t)pos * G4H + g) * BATCH +
                                mt[q] * 16 + hs * 8);
        if (tail) {
          const float wc = w0col[g];
#pragma unroll
          for (int r = 0; r < 8; ++r) {
            const int b = mt[q] * 16 + hs * 8 + r;
            const float d = predsh[b] - feat[((size_t)b * TLEN + pos) * 8];
            acc[gi][r] += d * wc;
          }
        }
      }
#pragma unroll 4
      for (int kb = 0; kb < KB0; ++kb) {
        const v16h a = load_a_lds(h0r, mt[q] * 16, kb * 32);
#pragma unroll
        for (int gi = 0; gi < 4; ++gi) {
          const v16h bm = load_b_pack(w0p, gi * 32 + ht[q], kb, KB0);
          acc[gi] = __builtin_amdgcn_wmma_f32_16x16x32_f16(
              false, a, false, bm, (short)0, acc[gi], false, false);
        }
      }
      // fused elementwise + writeback into the write buffer (no barrier needed)
#pragma unroll
      for (int r = 0; r < 8; ++r) {
        const float ig = sigmf(acc[0][r]);
        const float fg = sigmf(acc[1][r]);
        const float gg = tanhf(acc[2][r]);
        const float og = sigmf(acc[3][r]);
        const float c = fg * c0[q][r] + ig * gg;
        c0[q][r] = c;
        const float h = og * tanhf(c);
        const int b = mt[q] * 16 + hs * 8 + r;
        h0w[(size_t)b * HID + ht[q] * 16 + n] = (_Float16)h;
      }
    }
    __syncthreads();  // h0w complete; h0r/h1r reads may continue next phase

    // ======== layer 1: g = bias1 + [h0_new ; h1_prev] @ [Wih1;Whh1]^T ========
#pragma unroll 1
    for (int q = 0; q < 4; ++q) {
      v8f acc[4];
#pragma unroll
      for (int gi = 0; gi < 4; ++gi) {
        const float bz = bias1[gi * HID + ht[q] * 16 + n];
#pragma unroll
        for (int r = 0; r < 8; ++r) acc[gi][r] = bz;
      }
#pragma unroll 4
      for (int kb = 0; kb < KB1; ++kb) {
        const _Float16* abuf = (kb < KB0) ? h0w : h1r;
        const int k0 = (kb < KB0) ? kb * 32 : (kb - KB0) * 32;
        const v16h a = load_a_lds(abuf, mt[q] * 16, k0);
#pragma unroll
        for (int gi = 0; gi < 4; ++gi) {
          const v16h bm = load_b_pack(w1p, gi * 32 + ht[q], kb, KB1);
          acc[gi] = __builtin_amdgcn_wmma_f32_16x16x32_f16(
              false, a, false, bm, (short)0, acc[gi], false, false);
        }
      }
#pragma unroll
      for (int r = 0; r < 8; ++r) {
        const float ig = sigmf(acc[0][r]);
        const float fg = sigmf(acc[1][r]);
        const float gg = tanhf(acc[2][r]);
        const float og = sigmf(acc[3][r]);
        const float c = fg * c1[q][r] + ig * gg;
        c1[q][r] = c;
        const float h = og * tanhf(c);
        const int b = mt[q] * 16 + hs * 8 + r;
        h1w[(size_t)b * HID + ht[q] * 16 + n] = (_Float16)h;
      }
    }
    __syncthreads();  // h1w complete before next step swaps buffers
  }

  // ======== head: pred_b = h1_last[b] . Wout + bout ========
  if (tid < BATCH) predacc[tid] = 0.f;
  __syncthreads();
  {
    const int b = tid >> 4;
    const int seg = tid & 15;
    float p = 0.f;
#pragma unroll 4
    for (int j = seg * 32; j < seg * 32 + 32; ++j)
      p += (float)h1last[(size_t)b * HID + j] * Wout[j];
    atomicAdd(&predacc[b], p);
  }
  __syncthreads();
  if (tid < BATCH) {
    const float pv = predacc[tid] + bout[0];
    predbuf[(size_t)t * BATCH + tid] = pv;
    out[(size_t)tid * HOR + t] = pv;  // d_out layout [B, horizon, 1]
  }
  __threadfence();
  __syncthreads();
  if (tid == 0)
    __hip_atomic_store(&flags[t], 1, __ATOMIC_RELEASE, __HIP_MEMORY_SCOPE_AGENT);
}

extern "C" void kernel_launch(void* const* d_in, const int* in_sizes, int n_in,
                              void* d_out, int out_size, void* d_ws, size_t ws_size,
                              hipStream_t stream) {
  const float* feat = (const float*)d_in[0];
  const float* Wih0 = (const float*)d_in[1];
  const float* Whh0 = (const float*)d_in[2];
  const float* bih0 = (const float*)d_in[3];
  const float* bhh0 = (const float*)d_in[4];
  const float* Wih1 = (const float*)d_in[5];
  const float* Whh1 = (const float*)d_in[6];
  const float* bih1 = (const float*)d_in[7];
  const float* bhh1 = (const float*)d_in[8];
  const float* Wout = (const float*)d_in[9];
  const float* bout = (const float*)d_in[10];
  float* out = (float*)d_out;

  char* ws = (char*)d_ws;
  _Float16* w0p   = (_Float16*)(ws + OFF_W0P);
  _Float16* w1p   = (_Float16*)(ws + OFF_W1P);
  float*    bias1 = (float*)(ws + OFF_BIAS1);
  float*    w0col = (float*)(ws + OFF_W0COL);
  float*    pred  = (float*)(ws + OFF_PRED);
  int*      flags = (int*)(ws + OFF_FLAG);
  float*    Z0    = (float*)(ws + OFF_Z0);

  lstm_prep_pack_kernel<<<8192, 256, 0, stream>>>(Wih0, Whh0, Wih1, Whh1, bih1,
                                                  bhh1, w0p, w1p, bias1, w0col,
                                                  flags);
  lstm_zproj_kernel<<<16384, 256, 0, stream>>>(feat, Wih0, bih0, bhh0, Z0);
  lstm_window_kernel<<<HOR, 1024, 0, stream>>>(feat, Wout, bout, w0p, w1p,
                                               bias1, w0col, Z0, pred, flags,
                                               out);
}
